// MultiRelationalAttentionConv_17763984736779
// MI455X (gfx1250) — matrix operand order repaired and checked by reading, not compiled, not run on hardware
//
#include <hip/hip_runtime.h>

// ---------------------------------------------------------------------------
// MultiRelationalAttentionConv for MI455X (gfx1250, wave32, WMMA)
// Flash-style fused softmax(score * scale[col]) @ V.
//  - V pre-swizzled into WMMA B-fragment layout (contiguous v16h per lane)
//  - P computed directly in WMMA A-fragment register order (no LDS staging)
// ---------------------------------------------------------------------------

typedef __attribute__((ext_vector_type(16))) _Float16 v16h;
typedef __attribute__((ext_vector_type(8)))  float    v8f;

#define N_NODES 4096
#define DIMS    64
#define CHUNK   32
#define NCHUNK  (N_NODES / CHUNK)   // 128

union FragU {
    v16h     v;
    int4     q[2];
    _Float16 h[16];
};

// ---------------------------------------------------------------------------
// Kernel 1: per-node scale + V row ( = vsrc @ W^T + b ), V stored swizzled in
// WMMA B-fragment order:  flat = (((chunk*4 + ntile)*32) + h*16 + n)*16 + e
// where for element (j,d):  chunk=j/32, K=j%32, h=(K/8)%2, e=K%8+8*(K/16), ntile=d/16,
// n=d%16.  (ISA 7.12.2: B 32x16 f16, lane half h holds K=8h..8h+7, 16+8h..16+8h+7,
// N = lane%16, element e within v16h <-> K = 8h+e (e<8), 16+8h+e-8 (e>=8).)
// ---------------------------------------------------------------------------
__global__ __launch_bounds__(64) void prep_kernel(
    const float* __restrict__ embed,
    const float* __restrict__ deg_dev, const float* __restrict__ deg_task,
    const float* __restrict__ W1, const float* __restrict__ b1,
    const float* __restrict__ W2, const float* __restrict__ b2,
    const float* __restrict__ W3, const float* __restrict__ b3,
    float* __restrict__ scales, _Float16* __restrict__ vswz)
{
    const int rel = blockIdx.y;      // 0=collab, 1=interact, 2=similar
    const int i   = blockIdx.x;      // node
    const int d   = threadIdx.x;     // feature

    const float* dev  = embed;
    const float* task = embed + (size_t)N_NODES * DIMS;

    const float *src, *dst, *W, *b, *sdeg, *ddeg;
    if (rel == 0)      { src = dev;  dst = dev;  W = W1; b = b1; sdeg = deg_dev;  ddeg = deg_dev;  }
    else if (rel == 1) { src = dev;  dst = task; W = W2; b = b2; sdeg = deg_dev;  ddeg = deg_task; }
    else               { src = task; dst = task; W = W3; b = b3; sdeg = deg_task; ddeg = deg_task; }

    float sv   = src[(size_t)i * DIMS + d];
    float dv   = dst[(size_t)i * DIMS + d];
    float msgv = dv + sv * dv;                 // msg = dst + src*dst
    float vsrc = (rel == 2) ? dv : msgv;       // similar uses raw task rows for V

    __shared__ float red[DIMS];
    __shared__ float xv[DIMS];
    red[d] = sv * msgv;                        // dot(src, msg)
    xv[d]  = vsrc;
    __syncthreads();
    for (int off = 32; off > 0; off >>= 1) {
        if (d < off) red[d] += red[d + off];
        __syncthreads();
    }
    if (d == 0) {
        float dot = red[0];
        scales[rel * N_NODES + i] =
            dot / (sqrtf(sdeg[i] * ddeg[i] + 1e-8f) * 8.0f);   // sqrt(D)=8
    }

    // V[i][d] = b[d] + sum_k vsrc[k] * W[d][k]
    float acc = b[d];
    const float* wrow = W + d * DIMS;
    #pragma unroll 16
    for (int k = 0; k < DIMS; ++k) acc += xv[k] * wrow[k];

    // swizzled store
    int c = i >> 5, K = i & 31;
    int t = d >> 4, n = d & 15;
    int h = (K >> 3) & 1, e = (K & 7) | ((K >> 4) << 3);
    size_t idx = ((size_t)((c * 4 + t) * 32 + h * 16 + n)) * 16 + e;
    vswz[(size_t)rel * N_NODES * DIMS + idx] = (_Float16)acc;
}

// ---------------------------------------------------------------------------
// Kernel 2: fused online-softmax + P@V.  One wave32 per 16 output rows.
// Lane (m = lane%16, h = lane/16) owns row m with chunk columns in A-fragment
// order: K in [h*8, h*8+8) U [16+h*8, 16+h*8+8).  So p[] is directly the
// 16x32 f16 A fragment -- zero LDS traffic in the hot loop.
// ---------------------------------------------------------------------------
__global__ __launch_bounds__(32) void attn_kernel(
    const float* __restrict__ score0, const float* __restrict__ score1,
    const float* __restrict__ score2,
    const float* __restrict__ scales, const _Float16* __restrict__ vswz,
    float* __restrict__ O)
{
    const int rel = blockIdx.y;
    const float* __restrict__ score =
        (rel == 0) ? score0 : (rel == 1) ? score1 : score2;
    const float* __restrict__ scale = scales + rel * N_NODES;
    const v16h*  __restrict__ vb =
        ((const v16h*)vswz) + (size_t)rel * (NCHUNK * 4 * 32);
    float* __restrict__ out = O + (size_t)rel * N_NODES * DIMS;

    const int lane = threadIdx.x;
    const int m    = lane & 15;
    const int h    = lane >> 4;
    const int row0 = blockIdx.x * 16;

    v8f acc0 = {}, acc1 = {}, acc2 = {}, acc3 = {};
    float mrun = -1e30f;
    float lrun = 0.0f;

    // group-0 columns start at h*8, group-1 columns at 16+h*8 within a chunk
    const float* srow = score + (size_t)(row0 + m) * N_NODES;
    const int o0 = h * 8;
    const int o1 = 16 + h * 8;

    for (int c = 0; c < NCHUNK; ++c) {
        const int cb0 = c * CHUNK;
        // --- load 16 scores + 16 per-column scales in A-fragment order ---
        float4 sa0 = *(const float4*)(srow + cb0 + o0);
        float4 sa1 = *(const float4*)(srow + cb0 + o0 + 4);
        float4 sa2 = *(const float4*)(srow + cb0 + o1);
        float4 sa3 = *(const float4*)(srow + cb0 + o1 + 4);
        float4 ca0 = *(const float4*)(scale + cb0 + o0);
        float4 ca1 = *(const float4*)(scale + cb0 + o0 + 4);
        float4 ca2 = *(const float4*)(scale + cb0 + o1);
        float4 ca3 = *(const float4*)(scale + cb0 + o1 + 4);
        if (c + 1 < NCHUNK)
            __builtin_prefetch(srow + cb0 + CHUNK, 0, 1);

        float s[16];
        s[ 0] = sa0.x * ca0.x; s[ 1] = sa0.y * ca0.y;
        s[ 2] = sa0.z * ca0.z; s[ 3] = sa0.w * ca0.w;
        s[ 4] = sa1.x * ca1.x; s[ 5] = sa1.y * ca1.y;
        s[ 6] = sa1.z * ca1.z; s[ 7] = sa1.w * ca1.w;
        s[ 8] = sa2.x * ca2.x; s[ 9] = sa2.y * ca2.y;
        s[10] = sa2.z * ca2.z; s[11] = sa2.w * ca2.w;
        s[12] = sa3.x * ca3.x; s[13] = sa3.y * ca3.y;
        s[14] = sa3.z * ca3.z; s[15] = sa3.w * ca3.w;

        // --- online softmax stats (row m split across lanes m and m+16) ---
        float mx = s[0];
        #pragma unroll
        for (int k = 1; k < 16; ++k) mx = fmaxf(mx, s[k]);
        mx = fmaxf(mx, __shfl_xor(mx, 16, 32));
        float mnew = fmaxf(mrun, mx);

        // --- rescale accumulators only when some row's max advanced ---
        if (__ballot(mnew != mrun) != 0ull) {
            float f = __expf(mrun - mnew);
            #pragma unroll
            for (int r = 0; r < 8; ++r) {
                float fr = __shfl(f, h * 8 + r, 32);
                acc0[r] *= fr; acc1[r] *= fr; acc2[r] *= fr; acc3[r] *= fr;
            }
            lrun *= f;
        }
        mrun = mnew;

        // --- p = exp(s - m), already in A-fragment element order ---
        FragU ua;
        float ls = 0.0f;
        #pragma unroll
        for (int k = 0; k < 16; ++k) {
            float p = __expf(s[k] - mnew);
            ls += p;
            ua.h[k] = (_Float16)p;
        }
        ls  += __shfl_xor(ls, 16, 32);
        lrun += ls;

        // --- B fragments: contiguous pre-swizzled v16h per lane ---
        const int cb = (c * 4) * 32 + lane;
        v16h b0 = vb[cb];
        v16h b1 = vb[cb + 32];
        v16h b2 = vb[cb + 64];
        v16h b3 = vb[cb + 96];

        acc0 = __builtin_amdgcn_wmma_f32_16x16x32_f16(false, ua.v, false, b0,
                                                      (short)0, acc0, false, false);
        acc1 = __builtin_amdgcn_wmma_f32_16x16x32_f16(false, ua.v, false, b1,
                                                      (short)0, acc1, false, false);
        acc2 = __builtin_amdgcn_wmma_f32_16x16x32_f16(false, ua.v, false, b2,
                                                      (short)0, acc2, false, false);
        acc3 = __builtin_amdgcn_wmma_f32_16x16x32_f16(false, ua.v, false, b3,
                                                      (short)0, acc3, false, false);
    }

    // --- normalize by row sum and write O (C rows: VGPR r <-> M = r+h*8) ---
    #pragma unroll
    for (int r = 0; r < 8; ++r) {
        float lr  = __shfl(lrun, h * 8 + r, 32);
        float inv = 1.0f / lr;
        size_t base = (size_t)(row0 + h * 8 + r) * DIMS + m;
        out[base +  0] = acc0[r] * inv;
        out[base + 16] = acc1[r] * inv;
        out[base + 32] = acc2[r] * inv;
        out[base + 48] = acc3[r] * inv;
    }
}

// ---------------------------------------------------------------------------
// Kernel 3: out = leaky_relu((x) @ W3^T + b3, 0.2)
//   rows < N : x = dev  + O_collab + O_interact
//   rows >= N: x = task + O_similar
// ---------------------------------------------------------------------------
__global__ __launch_bounds__(64) void final_kernel(
    const float* __restrict__ embed, const float* __restrict__ O,
    const float* __restrict__ W3, const float* __restrict__ b3,
    float* __restrict__ outp)
{
    const int row = blockIdx.x;
    const int d   = threadIdx.x;
    __shared__ float xv[DIMS];

    float x;
    if (row < N_NODES) {
        x = embed[(size_t)row * DIMS + d]
          + O[(size_t)row * DIMS + d]
          + O[(size_t)N_NODES * DIMS + (size_t)row * DIMS + d];
    } else {
        int tr = row - N_NODES;
        x = embed[(size_t)row * DIMS + d]
          + O[(size_t)2 * N_NODES * DIMS + (size_t)tr * DIMS + d];
    }
    xv[d] = x;
    __syncthreads();

    float acc = b3[d];
    const float* wrow = W3 + d * DIMS;
    #pragma unroll 16
    for (int k = 0; k < DIMS; ++k) acc += xv[k] * wrow[k];

    outp[(size_t)row * DIMS + d] = (acc > 0.0f) ? acc : 0.2f * acc;
}

// ---------------------------------------------------------------------------
extern "C" void kernel_launch(void* const* d_in, const int* in_sizes, int n_in,
                              void* d_out, int out_size, void* d_ws, size_t ws_size,
                              hipStream_t stream)
{
    const float* embed    = (const float*)d_in[0];
    // d_in[1..3] = adjacency matrices (unused by the reference math)
    const float* cscore   = (const float*)d_in[4];
    const float* iscore   = (const float*)d_in[5];
    const float* sscore   = (const float*)d_in[6];
    const float* deg_dev  = (const float*)d_in[7];
    const float* deg_task = (const float*)d_in[8];
    const float* W1 = (const float*)d_in[9];  const float* b1 = (const float*)d_in[10];
    const float* W2 = (const float*)d_in[11]; const float* b2 = (const float*)d_in[12];
    const float* W3 = (const float*)d_in[13]; const float* b3 = (const float*)d_in[14];

    // workspace layout (base assumed >= 32B aligned):
    //   [0)                    vswz   : 3 * 4096*64 f16  = 1,572,864 B
    //   [1,572,864)            O      : 3 * 4096*64 f32  = 3,145,728 B
    //   [4,718,592)            scales : 3 * 4096    f32  =    49,152 B
    char* ws = (char*)d_ws;
    _Float16* vswz   = (_Float16*)ws;
    float*    O      = (float*)(ws + (size_t)3 * N_NODES * DIMS * 2);
    float*    scales = (float*)(ws + (size_t)3 * N_NODES * DIMS * 2
                                   + (size_t)3 * N_NODES * DIMS * 4);

    prep_kernel<<<dim3(N_NODES, 3), 64, 0, stream>>>(
        embed, deg_dev, deg_task, W1, b1, W2, b2, W3, b3, scales, vswz);

    attn_kernel<<<dim3(N_NODES / 16, 3), 32, 0, stream>>>(
        cscore, iscore, sscore, scales, vswz, O);

    final_kernel<<<2 * N_NODES, 64, 0, stream>>>(
        embed, O, W3, b3, (float*)d_out);
}
